// AttentionSAT_29265907155697
// MI455X (gfx1250) — compile-verified, matrix operand order checked by reading, not run on hardware
//
#include <hip/hip_runtime.h>
#include <hip/hip_bf16.h>
#include <math.h>

// ---------------------------------------------------------------------------
// NeuroSAT-style forward, 32 rounds, MI455X (gfx1250).
//  - GEMMs: v_wmma_f32_16x16x32_f16, one wave computes 1 M-tile x 4 N-tiles
//    (A k-block loaded once, 4 WMMAs against 4 weight panels -> 4x A reuse;
//    waves in a block share the A rows -> WGP$ broadcast).
//  - WGP-scope global_prefetch on the streamed A operand.
//  - Literal softmax/aggregation: CSR gather (built once per launch),
//    no float atomics in the round loop.
// ---------------------------------------------------------------------------

#define NVARS  32768
#define NCLS   131072
#define NLITS  65536
#define FF     256
#define F2     512
#define NNZ    (NCLS * 3)
#define NROUND 32

typedef __attribute__((ext_vector_type(16))) _Float16     v16h;
typedef __attribute__((ext_vector_type(8)))  float        v8f;
typedef __attribute__((ext_vector_type(8)))  unsigned int v8u;

__device__ __forceinline__ float softplusf(float x) {
  if (x > 20.f)  return x;
  if (x < -20.f) return expf(x);
  return log1pf(expf(x));
}

__device__ __forceinline__ int lit_of(int cl) {
  int var = ((cl > 0) ? cl : -cl) - 1;
  return (cl > 0) ? var : var + NVARS;
}

// ---------------------------------------------------------------------------
// WMMA GEMM: C[M,N] = epilogue(A[M,K] @ B[K,N] + bias)
//   A  : f16 row-major [M,K] (activations, streamed)
//   Bt : f16 [N,K] (weights pre-transposed; L2-resident)
//   One wave: 16 rows x 64 cols (4 accumulators). K-loop steps of 32.
//   Per-lane operand layout per CDNA5 ISA 7.12.2 (16-bit A 16x32):
//     lane = g*16 + r16; VGPR j holds K pair {kof[j], kof[j]+1},
//     kof[j] = (j<4 ? 2j : 16+2(j-4)) + 8g
// ---------------------------------------------------------------------------
__global__ void gemm_wmma_f16(const _Float16* __restrict__ A,
                              const _Float16* __restrict__ Bt,
                              const float*    __restrict__ bias,
                              float*          __restrict__ Cf,
                              _Float16*       __restrict__ Ch,
                              int M, int N, int K, int relu)
{
  const int lane = threadIdx.x & 31;
  const int wv   = threadIdx.x >> 5;
  const int tile = blockIdx.x * (blockDim.x >> 5) + wv;   // over (M/16)*(N/64)
  const int tilesN4 = N >> 6;
  const int tm  = tile / tilesN4;
  const int tn4 = tile - tm * tilesN4;
  if (tm * 16 >= M) return;

  const int r16 = lane & 15;   // A row-in-tile / B col-in-tile
  const int g   = lane >> 4;

  int kof[8];
#pragma unroll
  for (int j = 0; j < 8; ++j)
    kof[j] = ((j < 4) ? (2 * j) : (16 + 2 * (j - 4))) + g * 8;

  const _Float16* Arow  = A  + (size_t)(tm * 16 + r16) * K;
  const _Float16* Brow0 = Bt + (size_t)(tn4 * 64 + r16) * K;

  v8f acc0 = {}, acc1 = {}, acc2 = {}, acc3 = {};
  for (int k0 = 0; k0 < K; k0 += 32) {
    // WGP-scope speculative prefetch of the next A k-block (ISA 10.5:
    // scope 0 pulls into all cache levels; SYS/DEV would skip the WGP$).
    __builtin_prefetch(Arow + k0 + 32, 0, 3);
    v8u au;
#pragma unroll
    for (int j = 0; j < 8; ++j)
      au[j] = *(const unsigned int*)(Arow + k0 + kof[j]);
    const v16h a = __builtin_bit_cast(v16h, au);

    v8u bu0, bu1, bu2, bu3;
#pragma unroll
    for (int j = 0; j < 8; ++j) {
      bu0[j] = *(const unsigned int*)(Brow0 + (size_t)0 * 16 * K + k0 + kof[j]);
      bu1[j] = *(const unsigned int*)(Brow0 + (size_t)1 * 16 * K + k0 + kof[j]);
      bu2[j] = *(const unsigned int*)(Brow0 + (size_t)2 * 16 * K + k0 + kof[j]);
      bu3[j] = *(const unsigned int*)(Brow0 + (size_t)3 * 16 * K + k0 + kof[j]);
    }
    acc0 = __builtin_amdgcn_wmma_f32_16x16x32_f16(false, a, false,
             __builtin_bit_cast(v16h, bu0), (short)0, acc0, false, false);
    acc1 = __builtin_amdgcn_wmma_f32_16x16x32_f16(false, a, false,
             __builtin_bit_cast(v16h, bu1), (short)0, acc1, false, false);
    acc2 = __builtin_amdgcn_wmma_f32_16x16x32_f16(false, a, false,
             __builtin_bit_cast(v16h, bu2), (short)0, acc2, false, false);
    acc3 = __builtin_amdgcn_wmma_f32_16x16x32_f16(false, a, false,
             __builtin_bit_cast(v16h, bu3), (short)0, acc3, false, false);
  }

  v8f accs[4] = {acc0, acc1, acc2, acc3};
#pragma unroll
  for (int s = 0; s < 4; ++s) {
    const int col = tn4 * 64 + s * 16 + r16;
    const float bv = bias ? bias[col] : 0.f;
#pragma unroll
    for (int r = 0; r < 8; ++r) {
      const int row = tm * 16 + r + g * 8;    // C/D layout: VGPR r -> M = r + 8g
      float v = accs[s][r] + bv;
      if (relu) v = fmaxf(v, 0.f);
      const size_t o = (size_t)row * N + col;
      if (Cf) Cf[o] = v;
      if (Ch) Ch[o] = (_Float16)v;
    }
  }
}

// --------------------------- support kernels -------------------------------

__global__ void k_transpose_h(const float* __restrict__ W,
                              _Float16* __restrict__ Wt, int K, int N) {
  int idx = blockIdx.x * blockDim.x + threadIdx.x;
  if (idx >= K * N) return;
  int k = idx / N, n = idx - k * N;
  Wt[(size_t)n * K + k] = (_Float16)W[idx];
}

__global__ void k_zero_i(int* __restrict__ p, int n) {
  int i = blockIdx.x * blockDim.x + threadIdx.x;
  if (i < n) p[i] = 0;
}

__global__ void k_zero_f(float* __restrict__ p, int n) {
  int i = blockIdx.x * blockDim.x + threadIdx.x;
  if (i < n) p[i] = 0.f;
}

__global__ void k_initL(float* __restrict__ L, _Float16* __restrict__ Lh) {
  int idx = blockIdx.x * blockDim.x + threadIdx.x;   // NLITS*FF
  if (idx >= NLITS * FF) return;
  int f = idx & (FF - 1);
  float v = (((f == 0) ? 1.f : 0.f) - 1.f / (float)FF) * 4.f;  // sqrt(256)*0.25
  L[idx] = v;
  Lh[idx] = (_Float16)v;
}

// ---- CSR over literals (built once per launch) ----
__global__ void k_csr_count(const int* __restrict__ clauses, int* __restrict__ deg) {
  int e = blockIdx.x * blockDim.x + threadIdx.x;
  if (e >= NNZ) return;
  atomicAdd(&deg[lit_of(clauses[e])], 1);
}

// exclusive prefix sum over NLITS entries; single block of 1024 threads x 64 each
__global__ void k_csr_scan(const int* __restrict__ deg, int* __restrict__ start) {
  __shared__ int sb[1024];
  const int t = threadIdx.x;
  const int base = t * 64;
  int s = 0;
  for (int i = 0; i < 64; ++i) s += deg[base + i];
  sb[t] = s; __syncthreads();
  for (int o = 1; o < 1024; o <<= 1) {
    int v = (t >= o) ? sb[t - o] : 0;
    __syncthreads();
    sb[t] += v;
    __syncthreads();
  }
  int run = (t == 0) ? 0 : sb[t - 1];
  for (int i = 0; i < 64; ++i) { start[base + i] = run; run += deg[base + i]; }
}

__global__ void k_csr_fill(const int* __restrict__ clauses,
                           const int* __restrict__ start,
                           int* __restrict__ cursor, int* __restrict__ elist) {
  int e = blockIdx.x * blockDim.x + threadIdx.x;
  if (e >= NNZ) return;
  int lit = lit_of(clauses[e]);
  int slot = atomicAdd(&cursor[lit], 1);
  elist[start[lit] + slot] = e;
}

// ---- round kernels ----
__global__ void k_build_varh(const float* __restrict__ L,
                             _Float16* __restrict__ varh) {
  int idx = blockIdx.x * blockDim.x + threadIdx.x;   // NVARS*F2
  if (idx >= NVARS * F2) return;
  int v = idx >> 9, c = idx & (F2 - 1);
  float x = (c < FF) ? L[(size_t)v * FF + c]
                     : L[(size_t)(v + NVARS) * FF + (c - FF)];
  varh[idx] = (_Float16)x;
}

__global__ void k_memory(const int* __restrict__ clauses,
                         const float* __restrict__ q,
                         float* __restrict__ mem, _Float16* __restrict__ memh) {
  int idx = blockIdx.x * blockDim.x + threadIdx.x;   // NCLS*FF
  if (idx >= NCLS * FF) return;
  int c = idx >> 8, f = idx & (FF - 1);
  float acc = 0.f;
#pragma unroll
  for (int j = 0; j < 3; ++j) {
    int cl = clauses[c * 3 + j];
    int var = ((cl > 0) ? cl : -cl) - 1;
    float s = (cl > 0) ? 1.f : -1.f;
    acc += softplusf(q[(size_t)var * FF + f] * s);
  }
  float m = expf(-acc);
  mem[idx] = m;
  memh[idx] = (_Float16)m;
}

// one wave per edge: ew[e] = exp(tanh(qp[lit]+mp[clause]) . v)
__global__ void k_edge(const int* __restrict__ clauses,
                       const float* __restrict__ qp,
                       const float* __restrict__ mp,
                       const float* __restrict__ vvec,
                       float* __restrict__ ew) {
  int lane = threadIdx.x & 31;
  int wv = threadIdx.x >> 5;
  int e = blockIdx.x * (blockDim.x >> 5) + wv;
  if (e >= NNZ) return;
  int c = e / 3;
  int lit = lit_of(clauses[e]);
  const float* qr = qp + (size_t)lit * FF;
  const float* mr = mp + (size_t)c * FF;
  float s = 0.f;
  for (int f = lane; f < FF; f += 32)
    s += tanhf(qr[f] + mr[f]) * vvec[f];
  for (int o = 16; o; o >>= 1) s += __shfl_down(s, o, 32);
  if (lane == 0)
    ew[e] = expf(s);   // |score| <= ||v||_1 : safe in f32 without max-subtract
}

// block per literal, thread per feature: softmax-weighted sum of memory rows
__global__ void k_gather(const int* __restrict__ elist,
                         const int* __restrict__ start,
                         const int* __restrict__ deg,
                         const float* __restrict__ ew,
                         const float* __restrict__ mem,
                         float* __restrict__ newlits) {
  const int lit = blockIdx.x;
  const int t = threadIdx.x;             // 0..255
  const int s0 = start[lit], d = deg[lit];
  float dn = 0.f, acc = 0.f;
  for (int i = 0; i < d; ++i) {
    int e = elist[s0 + i];
    float w = ew[e];
    dn += w;
    acc += w * mem[(size_t)(e / 3) * FF + t];
  }
  newlits[(size_t)lit * FF + t] = (d > 0) ? acc / dn : 0.f;
}

__global__ void k_build_cat(const float* __restrict__ L,
                            const float* __restrict__ newlits,
                            _Float16* __restrict__ cat) {
  int idx = blockIdx.x * blockDim.x + threadIdx.x;   // NLITS*F2
  if (idx >= NLITS * F2) return;
  int r = idx >> 9, c = idx & (F2 - 1);
  float x = (c < FF) ? L[(size_t)r * FF + c]
                     : newlits[(size_t)r * FF + (c - FF)];
  cat[idx] = (_Float16)x;
}

__global__ void k_ln256(const float* __restrict__ X,
                        const float* __restrict__ gg, const float* __restrict__ bb,
                        float* __restrict__ Y, _Float16* __restrict__ Yh) {
  __shared__ float sb[FF];
  int r = blockIdx.x, t = threadIdx.x;
  float x = X[(size_t)r * FF + t];
  sb[t] = x; __syncthreads();
  for (int o = 128; o; o >>= 1) { if (t < o) sb[t] += sb[t + o]; __syncthreads(); }
  float mu = sb[0] / (float)FF; __syncthreads();
  float d = x - mu;
  sb[t] = d * d; __syncthreads();
  for (int o = 128; o; o >>= 1) { if (t < o) sb[t] += sb[t + o]; __syncthreads(); }
  float var = sb[0] / (float)FF;
  float y = d * rsqrtf(var + 1e-3f) * gg[t] + bb[t];
  size_t o = (size_t)r * FF + t;
  Y[o] = y;
  Yh[o] = (_Float16)y;
}

__global__ void k_outln(const float* __restrict__ L,
                        const float* __restrict__ gg, const float* __restrict__ bb,
                        _Float16* __restrict__ vh) {
  __shared__ float sb[F2];
  int v = blockIdx.x, t = threadIdx.x;
  float x = (t < FF) ? L[(size_t)v * FF + t]
                     : L[(size_t)(v + NVARS) * FF + (t - FF)];
  sb[t] = x; __syncthreads();
  for (int o = 256; o; o >>= 1) { if (t < o) sb[t] += sb[t + o]; __syncthreads(); }
  float mu = sb[0] / (float)F2; __syncthreads();
  float d = x - mu;
  sb[t] = d * d; __syncthreads();
  for (int o = 256; o; o >>= 1) { if (t < o) sb[t] += sb[t + o]; __syncthreads(); }
  float var = sb[0] / (float)F2;
  float y = d * rsqrtf(var + 1e-3f) * gg[t] + bb[t];
  vh[(size_t)v * F2 + t] = (_Float16)y;
}

__global__ void k_logits(const _Float16* __restrict__ o2,
                         const float* __restrict__ W2, const float* __restrict__ b2,
                         float* __restrict__ logits) {
  __shared__ float sb[256];
  int v = blockIdx.x, t = threadIdx.x;
  float s = (float)o2[(size_t)v * F2 + t] * W2[t] +
            (float)o2[(size_t)v * F2 + t + 256] * W2[t + 256];
  sb[t] = s; __syncthreads();
  for (int o = 128; o; o >>= 1) { if (t < o) sb[t] += sb[t + o]; __syncthreads(); }
  if (t == 0) logits[v] = sb[0] + b2[0];
}

__global__ void k_loss(const int* __restrict__ clauses,
                       const float* __restrict__ logits,
                       float* __restrict__ loss_acc) {
  int c = blockIdx.x * blockDim.x + threadIdx.x;
  if (c >= NCLS) return;
  float acc = 0.f;
#pragma unroll
  for (int j = 0; j < 3; ++j) {
    int cl = clauses[c * 3 + j];
    int var = ((cl > 0) ? cl : -cl) - 1;
    float s = (cl > 0) ? 1.f : -1.f;
    acc += softplusf(logits[var] * s);
  }
  float cval = expf(-acc);
  float l = logf(1.f - cval + 1e-10f);
  atomicAdd(loss_acc, l * l);
}

__global__ void k_finalize(const float* __restrict__ logits,
                           const float* __restrict__ loss_acc,
                           float* __restrict__ out) {
  int i = blockIdx.x * blockDim.x + threadIdx.x;
  if (i < NVARS) out[i] = logits[i];
  else if (i == NVARS) out[NVARS] = loss_acc[0] / (float)NROUND;
}

// ---------------------------------------------------------------------------
// Host driver
// ---------------------------------------------------------------------------
static inline int cdiv(long a, long b) { return (int)((a + b - 1) / b); }

extern "C" void kernel_launch(void* const* d_in, const int* in_sizes, int n_in,
                              void* d_out, int out_size, void* d_ws, size_t ws_size,
                              hipStream_t stream) {
  (void)in_sizes; (void)n_in; (void)out_size; (void)ws_size;

  // jax.tree_util leaf order (dict keys sorted at every level):
  const int*   clauses = (const int*)  d_in[0];    // [131072,3]
  //            n_vars  =                d_in[1]
  const float* att_v   = (const float*)d_in[2];    // [256,1]
  const float* att_wm  = (const float*)d_in[3];    // [256,256]
  const float* att_wq  = (const float*)d_in[4];    // [256,256]
  const float* litln_g = (const float*)d_in[5];
  const float* litln_b = (const float*)d_in[6];
  const float* litW0   = (const float*)d_in[7];    // [512,256]
  const float* litb0   = (const float*)d_in[8];
  const float* litW1   = (const float*)d_in[9];    // [256,256]
  const float* litb1   = (const float*)d_in[10];
  const float* litW2   = (const float*)d_in[11];   // [256,256]
  const float* litb2   = (const float*)d_in[12];
  const float* outln_g = (const float*)d_in[13];
  const float* outln_b = (const float*)d_in[14];
  const float* outW0   = (const float*)d_in[15];   // [512,512]
  const float* outb0   = (const float*)d_in[16];
  const float* outW1   = (const float*)d_in[17];   // [512,512]
  const float* outb1   = (const float*)d_in[18];
  const float* outW2   = (const float*)d_in[19];   // [512,1]
  const float* outb2   = (const float*)d_in[20];
  const float* qW0     = (const float*)d_in[21];   // [512,256]
  const float* qb0     = (const float*)d_in[22];
  const float* qW1     = (const float*)d_in[23];   // [256,256]
  const float* qb1     = (const float*)d_in[24];
  const float* qW2     = (const float*)d_in[25];   // [256,256]
  const float* qb2     = (const float*)d_in[26];
  //            rounds  =                d_in[27]

  float* out = (float*)d_out;

  // ---- workspace layout ----
  size_t off = 0;
  auto alloc = [&](size_t bytes) -> char* {
    char* p = (char*)d_ws + off;
    off += (bytes + 255) & ~(size_t)255;
    return p;
  };
  float*     L        = (float*)    alloc((size_t)NLITS * FF * 4);
  _Float16*  Lh       = (_Float16*) alloc((size_t)NLITS * FF * 2);
  _Float16*  varh     = (_Float16*) alloc((size_t)NVARS * F2 * 2);  // also vh (aliased)
  _Float16*  q1h      = (_Float16*) alloc((size_t)NVARS * FF * 2);
  _Float16*  q2h      = (_Float16*) alloc((size_t)NVARS * FF * 2);
  float*     q3f      = (float*)    alloc((size_t)NVARS * FF * 4);
  float*     memf     = (float*)    alloc((size_t)NCLS  * FF * 4);
  _Float16*  memh     = (_Float16*) alloc((size_t)NCLS  * FF * 2);
  float*     qpf      = (float*)    alloc((size_t)NLITS * FF * 4);  // reused as m3
  float*     mpf      = (float*)    alloc((size_t)NCLS  * FF * 4);  // reused for o1h/o2h
  float*     ew       = (float*)    alloc((size_t)NNZ * 4);
  float*     newlits  = (float*)    alloc((size_t)NLITS * FF * 4);
  _Float16*  cath     = (_Float16*) alloc((size_t)NLITS * F2 * 2);
  _Float16*  m1h      = (_Float16*) alloc((size_t)NLITS * FF * 2);
  _Float16*  m2h      = (_Float16*) alloc((size_t)NLITS * FF * 2);
  float*     logits   = (float*)    alloc((size_t)NVARS * 4);
  float*     loss_acc = (float*)    alloc(256);
  int*       deg      = (int*)      alloc((size_t)NLITS * 4);
  int*       startv   = (int*)      alloc((size_t)NLITS * 4);
  int*       cursor   = (int*)      alloc((size_t)NLITS * 4);
  int*       elist    = (int*)      alloc((size_t)NNZ * 4);
  _Float16*  qW0t     = (_Float16*) alloc((size_t)F2 * FF * 2);
  _Float16*  qW1t     = (_Float16*) alloc((size_t)FF * FF * 2);
  _Float16*  qW2t     = (_Float16*) alloc((size_t)FF * FF * 2);
  _Float16*  litW0t   = (_Float16*) alloc((size_t)F2 * FF * 2);
  _Float16*  litW1t   = (_Float16*) alloc((size_t)FF * FF * 2);
  _Float16*  litW2t   = (_Float16*) alloc((size_t)FF * FF * 2);
  _Float16*  wqT      = (_Float16*) alloc((size_t)FF * FF * 2);
  _Float16*  wmT      = (_Float16*) alloc((size_t)FF * FF * 2);
  _Float16*  oW0t     = (_Float16*) alloc((size_t)F2 * F2 * 2);
  _Float16*  oW1t     = (_Float16*) alloc((size_t)F2 * F2 * 2);
  _Float16*  o1h      = (_Float16*)mpf;                      // alias (mp dead post-edge)
  _Float16*  o2h      = (_Float16*)((char*)mpf + (size_t)NVARS * F2 * 2);
  float*     m3f      = qpf;                                 // alias (qp dead post-edge)
  _Float16*  vh       = varh;                                // alias (varh dead post-query)

  const int TB = 256;

  // ---- one-time per launch: weights, CSR, init state ----
  k_transpose_h<<<cdiv((long)F2 * FF, TB), TB, 0, stream>>>(qW0, qW0t, F2, FF);
  k_transpose_h<<<cdiv((long)FF * FF, TB), TB, 0, stream>>>(qW1, qW1t, FF, FF);
  k_transpose_h<<<cdiv((long)FF * FF, TB), TB, 0, stream>>>(qW2, qW2t, FF, FF);
  k_transpose_h<<<cdiv((long)F2 * FF, TB), TB, 0, stream>>>(litW0, litW0t, F2, FF);
  k_transpose_h<<<cdiv((long)FF * FF, TB), TB, 0, stream>>>(litW1, litW1t, FF, FF);
  k_transpose_h<<<cdiv((long)FF * FF, TB), TB, 0, stream>>>(litW2, litW2t, FF, FF);
  k_transpose_h<<<cdiv((long)FF * FF, TB), TB, 0, stream>>>(att_wq, wqT, FF, FF);
  k_transpose_h<<<cdiv((long)FF * FF, TB), TB, 0, stream>>>(att_wm, wmT, FF, FF);
  k_transpose_h<<<cdiv((long)F2 * F2, TB), TB, 0, stream>>>(outW0, oW0t, F2, F2);
  k_transpose_h<<<cdiv((long)F2 * F2, TB), TB, 0, stream>>>(outW1, oW1t, F2, F2);

  k_zero_i<<<cdiv((long)NLITS, TB), TB, 0, stream>>>(deg, NLITS);
  k_zero_i<<<cdiv((long)NLITS, TB), TB, 0, stream>>>(cursor, NLITS);
  k_csr_count<<<cdiv((long)NNZ, TB), TB, 0, stream>>>(clauses, deg);
  k_csr_scan<<<1, 1024, 0, stream>>>(deg, startv);
  k_csr_fill<<<cdiv((long)NNZ, TB), TB, 0, stream>>>(clauses, startv, cursor, elist);

  k_initL<<<cdiv((long)NLITS * FF, TB), TB, 0, stream>>>(L, Lh);
  k_zero_f<<<1, 64, 0, stream>>>(loss_acc, 1);

  // 8 waves per block; consecutive waves share A rows (same tm) -> WGP$ broadcast
  auto gemm = [&](const _Float16* A, const _Float16* Bt, const float* bias,
                  float* Cf, _Float16* Ch, int M, int N, int K, int relu) {
    int tiles = (M / 16) * (N / 64);
    gemm_wmma_f16<<<tiles / 8, 256, 0, stream>>>(A, Bt, bias, Cf, Ch, M, N, K, relu);
  };

  for (int rd = 0; rd < NROUND; ++rd) {
    // query = mlp([L+, L-])
    k_build_varh<<<cdiv((long)NVARS * F2, TB), TB, 0, stream>>>(L, varh);
    gemm(varh, qW0t, qb0, nullptr, q1h, NVARS, FF, F2, 1);
    gemm(q1h,  qW1t, qb1, nullptr, q2h, NVARS, FF, FF, 1);
    gemm(q2h,  qW2t, qb2, q3f, nullptr, NVARS, FF, FF, 0);

    // memory = exp(-sum softplus(lit vals))
    k_memory<<<cdiv((long)NCLS * FF, TB), TB, 0, stream>>>(clauses, q3f, memf, memh);

    // additive attention projections
    gemm(Lh,   wqT, nullptr, qpf, nullptr, NLITS, FF, FF, 0);
    gemm(memh, wmT, nullptr, mpf, nullptr, NCLS,  FF, FF, 0);

    // per-edge scores, then CSR gather = softmax-weighted memory aggregation
    k_edge<<<cdiv((long)NNZ, 4), 128, 0, stream>>>(clauses, qpf, mpf, att_v, ew);
    k_gather<<<NLITS, FF, 0, stream>>>(elist, startv, deg, ew, memf, newlits);

    // L = LN(mlp([L, new_lits]))
    k_build_cat<<<cdiv((long)NLITS * F2, TB), TB, 0, stream>>>(L, newlits, cath);
    gemm(cath, litW0t, litb0, nullptr, m1h, NLITS, FF, F2, 1);
    gemm(m1h,  litW1t, litb1, nullptr, m2h, NLITS, FF, FF, 1);
    gemm(m2h,  litW2t, litb2, m3f, nullptr, NLITS, FF, FF, 0);
    k_ln256<<<NLITS, FF, 0, stream>>>(m3f, litln_g, litln_b, L, Lh);

    // logits = out_mlp(LN([L+, L-]))
    k_outln<<<NVARS, F2, 0, stream>>>(L, outln_g, outln_b, vh);
    gemm(vh,  oW0t, outb0, nullptr, o1h, NVARS, F2, F2, 1);
    gemm(o1h, oW1t, outb1, nullptr, o2h, NVARS, F2, F2, 1);
    k_logits<<<NVARS, 256, 0, stream>>>(o2h, outW2, outb2, logits);

    // loss accumulation (all rounds, mean at the end)
    k_loss<<<cdiv((long)NCLS, TB), TB, 0, stream>>>(clauses, logits, loss_acc);
  }

  k_finalize<<<cdiv((long)NVARS + 1, TB), TB, 0, stream>>>(logits, loss_acc, out);
}
